// Block_2302102470917
// MI455X (gfx1250) — matrix operand order repaired and checked by reading, not compiled, not run on hardware
//
#include <hip/hip_runtime.h>
#include <cstdint>
#include <cstddef>

// ---------------- problem dims ----------------
#define BB   2
#define SL   2048
#define DD   768
#define KK   16
#define KUU  3
#define NN   4096
#define NF   2049          // NN/2+1
#define HH   3072
#define HH2  6144
#define RR   4096          // BB*SL

typedef __attribute__((ext_vector_type(16))) __bf16 v16bf;
typedef __attribute__((ext_vector_type(8)))  float  v8f;
typedef __attribute__((ext_vector_type(4)))  unsigned int v4u;
typedef __attribute__((ext_vector_type(8)))  int v8i_;
typedef __attribute__((ext_vector_type(4)))  int v4i_;

#define LDT 40             // LDS tile row stride in halves (80B; 16B-aligned chunks)
#define TSZ (64 * LDT)     // one 64-row tile in halves

// ---------------- WMMA helpers ----------------
static __device__ inline v8f wmma_bf(v16bf a, v16bf b, v8f c) {
  return __builtin_amdgcn_wmma_f32_16x16x32_bf16(false, a, false, b, (short)0, c, false, false);
}

// A fragment: 16x32 bf16 tile, row-major in LDS (stride LDT halves).
static __device__ inline v16bf load_a_frag(const __bf16* At, int lane) {
  int row = lane & 15, hi = lane >> 4;
  const __bf16* p0 = At + row * LDT + hi * 8;
  v16bf a;
#pragma unroll
  for (int j = 0; j < 8; ++j) { a[j] = p0[j]; a[j + 8] = p0[16 + j]; }
  return a;
}

// B fragment: 32x16 tile stored k-contiguous per column: Bt[col][k].
static __device__ inline v16bf load_b_frag(const __bf16* Bt, int lane) {
  int col = lane & 15, kh = lane >> 4;
  const __bf16* p = Bt + col * LDT + kh * 8;
  v16bf b;
#pragma unroll
  for (int j = 0; j < 8; ++j) { b[j] = p[j]; b[j + 8] = p[16 + j]; }
  return b;
}

static __device__ inline v16bf splat16(float v) {
  __bf16 b = (__bf16)v;
  v16bf r;
#pragma unroll
  for (int j = 0; j < 16; ++j) r[j] = b;
  return r;
}

// ---------------- TDM: load a 64x32 bf16 tile into a padded LDS tile ----------------
// Global: 64 rows of 32 elems, row stride = stride_elems (2B elems).
// LDS: rows padded to LDT(=40) halves -> pad 4 DWORDs after every 16 DWORDs stored.
static __device__ inline void tdm_load_64x32(unsigned lds_byte, const void* gptr,
                                             unsigned stride_elems) {
  size_t ga = (size_t)gptr;
  v4u g0;
  g0[0] = 1u;                                               // count=1, user D#
  g0[1] = lds_byte;                                         // lds_addr (bytes)
  g0[2] = (unsigned)(ga & 0xFFFFFFFFu);                     // global_addr[31:0]
  g0[3] = (unsigned)((ga >> 32) & 0x1FFFFFFu) | 0x80000000u; // global_addr[56:32] | type=2
  unsigned w0 = (1u << 16)    // data_size = 2 bytes
              | (1u << 20)    // pad_enable
              | (3u << 22)    // pad_interval: 16 dwords
              | (3u << 25);   // pad_amount: 4 dwords
  unsigned dim0 = (unsigned)DD;     // tensor_dim0 (no OOB expected)
  unsigned dim1 = 1u << 20;         // tensor_dim1 (large)
  unsigned til0 = 32u, til1 = 64u;
  v8i_ g1;
  g1[0] = (int)w0;                                           // mask=0 | flags
  g1[1] = (int)((dim0 & 0xFFFFu) << 16);                     // abar_addr=0, dim0[15:0]
  g1[2] = (int)(((dim0 >> 16) & 0xFFFFu) | ((dim1 & 0xFFFFu) << 16));
  g1[3] = (int)(((dim1 >> 16) & 0xFFFFu) | (til0 << 16));    // dim1 hi | tile_dim0
  g1[4] = (int)til1;                                         // tile_dim1, tile_dim2=0
  g1[5] = (int)stride_elems;                                 // dim0_stride[31:0]
  g1[6] = 0;                                                 // dim0_stride[47:32], dim1_stride lo
  g1[7] = 0;
  v4i_ zz; zz[0] = zz[1] = zz[2] = zz[3] = 0;
  v8i_ z8;
#pragma unroll
  for (int j = 0; j < 8; ++j) z8[j] = 0;
  __builtin_amdgcn_tensor_load_to_lds(g0, g1, zz, zz, z8, 0);
}

// ---------------- workspace layout ----------------
static constexpr size_t SZ_XN1F = (size_t)RR * DD * 4;
static constexpr size_t SZ_XN1B = (size_t)RR * DD * 2;
static constexpr size_t SZ_U    = (size_t)BB * 2 * NF * DD * 2;     // bf16 re/im planes
static constexpr size_t SZ_SPEC = (size_t)RR * DD * 4;
static constexpr size_t SZ_OUT1 = (size_t)RR * DD * 4;
static constexpr size_t SZ_XN2  = (size_t)RR * DD * 2;
static constexpr size_t SZ_MPT  = (size_t)KK * DD * DD * 2;
static constexpr size_t SZ_MMT  = (size_t)KK * DD * DD * 2;
static constexpr size_t SZ_MUT  = (size_t)KUU * DD * DD * 2;
static constexpr size_t SZ_FC1B = (size_t)DD * HH2 * 2;
static constexpr size_t SZ_FC2B = (size_t)HH * DD * 2;
static constexpr size_t SZ_Z    = (size_t)BB * 2 * 2 * NF * DD * 4; // f32 (b,m,part,f,d)
static constexpr size_t SZ_H    = (size_t)RR * HH2 * 2;

static constexpr size_t OFS_XN1F = 0;
static constexpr size_t OFS_XN1B = OFS_XN1F + SZ_XN1F;
static constexpr size_t OFS_U    = OFS_XN1B + SZ_XN1B;
static constexpr size_t OFS_SPEC = OFS_U + SZ_U;
static constexpr size_t OFS_OUT1 = OFS_SPEC + SZ_SPEC;
static constexpr size_t OFS_XN2  = OFS_OUT1 + SZ_OUT1;
static constexpr size_t OFS_MPT  = OFS_XN2 + SZ_XN2;
static constexpr size_t OFS_MMT  = OFS_MPT + SZ_MPT;
static constexpr size_t OFS_MUT  = OFS_MMT + SZ_MMT;
static constexpr size_t OFS_FC1B = OFS_MUT + SZ_MUT;
static constexpr size_t OFS_FC2B = OFS_FC1B + SZ_FC1B;
static constexpr size_t OFS_BIG  = OFS_FC2B + SZ_FC2B;
static constexpr size_t OFS_Z    = OFS_BIG;            // dead after fft_inv
static constexpr size_t OFS_H    = OFS_BIG;            // aliases Z (used after)
static constexpr size_t OFS_ACT  = OFS_BIG + (SZ_Z > SZ_H ? SZ_Z : SZ_H);

// ---------------- prep kernels ----------------
__global__ void prep_phi_kernel(const float* __restrict__ M, const float* __restrict__ sigma,
                                __bf16* __restrict__ out) {
  size_t idx = (size_t)blockIdx.x * 256 + threadIdx.x;
  if (idx >= (size_t)KK * DD * DD) return;
  int k = (int)(idx / (DD * DD));
  float sc = powf(sigma[k], 0.25f);
  out[idx] = (__bf16)(M[idx] * sc);
}

__global__ void prep_mu_kernel(const float* __restrict__ M, __bf16* __restrict__ out) {
  size_t idx = (size_t)blockIdx.x * 256 + threadIdx.x;
  if (idx >= (size_t)KUU * DD * DD) return;
  out[idx] = (__bf16)M[idx];
}

// tiled transpose + bf16 cast: src (R x C) -> dst (C x R)
__global__ __launch_bounds__(256) void transpose_bf_kernel(const float* __restrict__ src,
                                                           __bf16* __restrict__ dst,
                                                           int R, int C) {
  __shared__ __bf16 t[32][33];
  int c0 = blockIdx.x * 32, r0 = blockIdx.y * 32;
  int tx = threadIdx.x & 31, ty = threadIdx.x >> 5;
#pragma unroll
  for (int j = 0; j < 4; ++j) {
    int r = r0 + ty + j * 8;
    t[ty + j * 8][tx] = (__bf16)src[(size_t)r * C + c0 + tx];
  }
  __syncthreads();
#pragma unroll
  for (int j = 0; j < 4; ++j) {
    int c = c0 + ty + j * 8;
    dst[(size_t)c * R + r0 + tx] = t[tx][ty + j * 8];
  }
}

// ---------------- RMSNorm ----------------
__global__ __launch_bounds__(256) void rmsnorm_kernel(const float* __restrict__ in,
                                                      const float* __restrict__ w,
                                                      float* __restrict__ outF,
                                                      __bf16* __restrict__ outB) {
  __shared__ float red[256];
  int r = blockIdx.x;
  int tid = threadIdx.x;
  const float* row = in + (size_t)r * DD;
  float v0 = row[tid], v1 = row[tid + 256], v2 = row[tid + 512];
  red[tid] = v0 * v0 + v1 * v1 + v2 * v2;
  __syncthreads();
  for (int off = 128; off > 0; off >>= 1) {
    if (tid < off) red[tid] += red[tid + off];
    __syncthreads();
  }
  float scale = rsqrtf(red[0] * (1.0f / DD) + 1e-5f);
#pragma unroll
  for (int j = 0; j < 3; ++j) {
    int c = tid + j * 256;
    float y = row[c] * scale * w[c];
    if (outF) outF[(size_t)r * DD + c] = y;
    outB[(size_t)r * DD + c] = (__bf16)y;
  }
}

// ---------------- 4096-pt in-place radix-2 DIT FFT in LDS ----------------
static __device__ void fft_stages(float2* buf, int tid, float sgn) {
  for (int s = 1; s <= 12; ++s) {
    int half = 1 << (s - 1);
    float base = sgn * 6.283185307179586f / (float)(1 << s);
#pragma unroll
    for (int q = 0; q < 8; ++q) {
      int bi = q * 256 + tid;
      int g = bi >> (s - 1);
      int p = bi & (half - 1);
      int i0 = (g << s) + p;
      int i1 = i0 + half;
      float ang = base * (float)p;
      float sv, cv;
      __sincosf(ang, &sv, &cv);
      float2 a = buf[i0];
      float2 bb = buf[i1];
      float tx = bb.x * cv - bb.y * sv;
      float ty = bb.x * sv + bb.y * cv;
      buf[i0].x = a.x + tx; buf[i0].y = a.y + ty;
      buf[i1].x = a.x - tx; buf[i1].y = a.y - ty;
    }
    __syncthreads();
  }
}

// Forward rfft of two packed real channels (d0,d1), zero-padded 2048->4096.
__global__ __launch_bounds__(256) void fft_fwd_kernel(const float* __restrict__ xn,
                                                      __bf16* __restrict__ U) {
  __shared__ float2 buf[NN];
  int tid = threadIdx.x;
  int c = blockIdx.x;
  int b = blockIdx.y;
  int d0 = 2 * c, d1 = d0 + 1;
#pragma unroll
  for (int jj = 0; jj < 16; ++jj) {
    int t = jj * 256 + tid;
    float re = 0.f, im = 0.f;
    if (t < SL) {
      re = xn[((size_t)b * SL + t) * DD + d0];
      im = xn[((size_t)b * SL + t) * DD + d1];
    }
    int rv = __brev((unsigned)t) >> 20;
    buf[rv].x = re; buf[rv].y = im;
  }
  __syncthreads();
  fft_stages(buf, tid, -1.0f);
  size_t pre = (size_t)(b * 2 + 0) * NF * DD;
  size_t pim = (size_t)(b * 2 + 1) * NF * DD;
#pragma unroll
  for (int jj = 0; jj < 9; ++jj) {
    int f = jj * 256 + tid;
    if (f <= 2048) {
      float2 a = buf[f];
      float2 bb = buf[(NN - f) & (NN - 1)];
      size_t o = (size_t)f * DD;
      U[pre + o + d0] = (__bf16)(0.5f * (a.x + bb.x));
      U[pim + o + d0] = (__bf16)(0.5f * (a.y - bb.y));
      U[pre + o + d1] = (__bf16)(0.5f * (a.y + bb.y));
      U[pim + o + d1] = (__bf16)(0.5f * (bb.x - a.x));
    }
  }
}

// ---------------- frequency-domain contraction (the big WMMA kernel) ----------------
__global__ __launch_bounds__(128) void contract_kernel(const __bf16* __restrict__ U,
                                                       const __bf16* __restrict__ Mp,
                                                       const __bf16* __restrict__ Mm,
                                                       const float* __restrict__ V,
                                                       float* __restrict__ Z) {
  __shared__ __bf16 Ar[TSZ];
  __shared__ __bf16 Ai[TSZ];
  __shared__ __bf16 Btd[2 * TSZ];
  int tid = threadIdx.x, lane = tid & 31, w = tid >> 5;
  int f0 = blockIdx.x * 64, d0 = blockIdx.y * 64;
  int bm = blockIdx.z;
  int b = bm >> 1, m = bm & 1;
  const __bf16* M = m ? Mm : Mp;
  const size_t planeRe = (size_t)(b * 2 + 0) * NF * DD;
  const size_t planeIm = (size_t)(b * 2 + 1) * NF * DD;

  float vr_[KK], vi_[KK];
  {
    int frow = f0 + w * 16 + (lane & 15);
    bool fok = (frow <= 2048);
    int fcl2 = fok ? frow : 0;
    const float* vrow = V + (size_t)fcl2 * KK * 2;
#pragma unroll
    for (int k = 0; k < KK; ++k) {
      vr_[k] = fok ? vrow[2 * k] : 0.f;
      vi_[k] = fok ? vrow[2 * k + 1] : 0.f;
    }
  }

  v8f zero = {0.f, 0.f, 0.f, 0.f, 0.f, 0.f, 0.f, 0.f};
  v8f zre[4], zim[4];
#pragma unroll
  for (int i = 0; i < 4; ++i) { zre[i] = zero; zim[i] = zero; }

  int rr = tid >> 1, ch = (tid & 1) * 16;
  int f = f0 + rr;
  int fs = m ? (2048 - f) : f;
  bool ok = (fs >= 0) && (fs <= 2048);
  int fcl = ok ? fs : 0;
  unsigned fm = m ? 0x80008000u : 0u;
  uint4 z4; z4.x = z4.y = z4.z = z4.w = 0u;

  for (int i0 = 0; i0 < DD; i0 += 32) {
    { // stage A (once per i0)
      const uint4* pre4 = (const uint4*)(U + planeRe + (size_t)fcl * DD + i0 + ch);
      const uint4* pim4 = (const uint4*)(U + planeIm + (size_t)fcl * DD + i0 + ch);
      uint4 a0 = pre4[0], a1 = pre4[1];
      uint4 b0 = pim4[0], b1 = pim4[1];
      b0.x ^= fm; b0.y ^= fm; b0.z ^= fm; b0.w ^= fm;
      b1.x ^= fm; b1.y ^= fm; b1.z ^= fm; b1.w ^= fm;
      if (!ok) { a0 = z4; a1 = z4; b0 = z4; b1 = z4; }
      uint4* ad = (uint4*)(Ar + rr * LDT + ch);
      uint4* bd = (uint4*)(Ai + rr * LDT + ch);
      ad[0] = a0; ad[1] = a1;
      bd[0] = b0; bd[1] = b1;
    }
    { // stage B for k=0 into buffer 0
      const uint4* bs = (const uint4*)(M + ((size_t)0 * DD + d0 + rr) * DD + i0 + ch);
      uint4* bd = (uint4*)(Btd + rr * LDT + ch);
      bd[0] = bs[0]; bd[1] = bs[1];
    }
    __syncthreads();
    v16bf araw = load_a_frag(Ar + w * 16 * LDT, lane);
    v16bf airaw = load_a_frag(Ai + w * 16 * LDT, lane);
#pragma unroll
    for (int k = 0; k < KK; ++k) {
      const __bf16* Bcur = Btd + (k & 1) * TSZ;
      if (k + 1 < KK) {
        const uint4* bs = (const uint4*)(M + ((size_t)(k + 1) * DD + d0 + rr) * DD + i0 + ch);
        uint4* bd = (uint4*)(Btd + ((k + 1) & 1) * TSZ + rr * LDT + ch);
        bd[0] = bs[0]; bd[1] = bs[1];
      }
      v16bf vr16 = splat16(vr_[k]);
      v16bf vi16 = splat16(vi_[k]);
      v16bf arp = vr16 * araw - vi16 * airaw;
      v16bf aip = vr16 * airaw + vi16 * araw;
#pragma unroll
      for (int ns = 0; ns < 4; ++ns) {
        v16bf bf = load_b_frag(Bcur + ns * 16 * LDT, lane);
        zre[ns] = wmma_bf(arp, bf, zre[ns]);
        zim[ns] = wmma_bf(aip, bf, zim[ns]);
      }
      __syncthreads();
    }
  }
  int hi = lane >> 4, col = lane & 15;
  const size_t PL = (size_t)NF * DD;
  size_t zbase = (size_t)((b * 2 + m) * 2) * PL;
#pragma unroll
  for (int ns = 0; ns < 4; ++ns)
#pragma unroll
    for (int v = 0; v < 8; ++v) {
      int ff = f0 + w * 16 + hi * 8 + v;
      if (ff <= 2048) {
        size_t o = (size_t)ff * DD + d0 + ns * 16 + col;
        Z[zbase + o] = zre[ns][v];
        Z[zbase + PL + o] = zim[ns][v];
      }
    }
}

// ---------------- inverse rfft (packed pair) + plus/minus combine ----------------
__global__ __launch_bounds__(256) void fft_inv_kernel(const float* __restrict__ Z,
                                                      float* __restrict__ spec) {
  __shared__ float2 buf[NN];
  int tid = threadIdx.x;
  int c = blockIdx.x, b = blockIdx.y;
  int d0 = 2 * c, d1 = d0 + 1;
  float p0[8], p1[8];
  const size_t PL = (size_t)NF * DD;
  for (int m = 0; m < 2; ++m) {
    __syncthreads();
#pragma unroll
    for (int jj = 0; jj < 9; ++jj) {
      int f = jj * 256 + tid;
      if (f <= 2048) {
        size_t zb = (size_t)((b * 2 + m) * 2) * PL;
        size_t o = (size_t)f * DD;
        float z0r = Z[zb + o + d0];
        float z0i = Z[zb + PL + o + d0];
        float z1r = Z[zb + o + d1];
        float z1i = Z[zb + PL + o + d1];
        int rv = __brev((unsigned)f) >> 20;
        buf[rv].x = z0r - z1i; buf[rv].y = z0i + z1r;
        if (f >= 1 && f <= 2047) {
          int rv2 = __brev((unsigned)(NN - f)) >> 20;
          buf[rv2].x = z0r + z1i; buf[rv2].y = z1r - z0i;
        }
      }
    }
    __syncthreads();
    fft_stages(buf, tid, +1.0f);
    if (m == 0) {
#pragma unroll
      for (int jj = 0; jj < 8; ++jj) {
        int t = jj * 256 + tid;
        p0[jj] = buf[t].x; p1[jj] = buf[t].y;
      }
    } else {
      const float inv = 1.0f / (float)NN;
#pragma unroll
      for (int jj = 0; jj < 8; ++jj) {
        int t = jj * 256 + tid;
        float altv = (t & 1) ? -1.f : 1.f;
        spec[((size_t)b * SL + t) * DD + d0] = (p0[jj] + altv * buf[t].x) * inv;
        spec[((size_t)b * SL + t) * DD + d1] = (p1[jj] + altv * buf[t].y) * inv;
      }
    }
  }
}

// ---------------- AR GEMM (3 shifted matmuls) + residual(x + spec) ----------------
__global__ __launch_bounds__(128) void ar_kernel(const __bf16* __restrict__ xnb,
                                                 const __bf16* __restrict__ Mut,
                                                 const float* __restrict__ x,
                                                 const float* __restrict__ spec,
                                                 float* __restrict__ out1) {
  __shared__ __bf16 At[2][TSZ];
  __shared__ __bf16 Bt[2][TSZ];
  int tid = threadIdx.x, lane = tid & 31, w = tid >> 5;
  int r0 = blockIdx.x * 64, n0 = blockIdx.y * 64;
  v8f zero = {0.f, 0.f, 0.f, 0.f, 0.f, 0.f, 0.f, 0.f};
  v8f acc[4];
#pragma unroll
  for (int i = 0; i < 4; ++i) acc[i] = zero;
  int rr = tid >> 1, ch = (tid & 1) * 16;
  int grow = r0 + rr;
  int bbb = grow >> 11, s = grow & 2047;
  uint4 z4; z4.x = z4.y = z4.z = z4.w = 0u;

  auto stage = [&](int it, int par) {
    int sh = it / 24;
    int i0 = (it % 24) * 32;
    int ss = s - sh;
    bool ok = ss >= 0;
    int scl = ok ? ss : 0;
    const uint4* as = (const uint4*)(xnb + ((size_t)(bbb << 11) + scl) * DD + i0 + ch);
    uint4 a0 = as[0], a1 = as[1];
    if (!ok) { a0 = z4; a1 = z4; }
    uint4* ad = (uint4*)(At[par] + rr * LDT + ch);
    ad[0] = a0; ad[1] = a1;
    const uint4* bs = (const uint4*)(Mut + ((size_t)sh * DD + n0 + rr) * DD + i0 + ch);
    uint4* bd = (uint4*)(Bt[par] + rr * LDT + ch);
    bd[0] = bs[0]; bd[1] = bs[1];
  };

  stage(0, 0);
  __syncthreads();
  for (int it = 0; it < KUU * 24; ++it) {
    int par = it & 1;
    if (it + 1 < KUU * 24) stage(it + 1, par ^ 1);
    v16bf af = load_a_frag(At[par] + w * 16 * LDT, lane);
#pragma unroll
    for (int ns = 0; ns < 4; ++ns) {
      v16bf bf = load_b_frag(Bt[par] + ns * 16 * LDT, lane);
      acc[ns] = wmma_bf(af, bf, acc[ns]);
    }
    __syncthreads();
  }
  int hi = lane >> 4, col = lane & 15;
#pragma unroll
  for (int ns = 0; ns < 4; ++ns)
#pragma unroll
    for (int v = 0; v < 8; ++v) {
      int gr = r0 + w * 16 + hi * 8 + v;
      int gc = n0 + ns * 16 + col;
      size_t o = (size_t)gr * DD + gc;
      out1[o] = x[o] + spec[o] + acc[ns][v];
    }
}

// ---------------- fc1 GEMM via async global->LDS loads (ASYNCcnt path) ----------------
__global__ __launch_bounds__(128) void fc1_kernel(const __bf16* __restrict__ A,
                                                  const __bf16* __restrict__ Bw,  // (n,k) 6144x768
                                                  __bf16* __restrict__ Hout) {
  __shared__ __bf16 At[2][TSZ];
  __shared__ __bf16 Bt[2][TSZ];
  int tid = threadIdx.x, lane = tid & 31, w = tid >> 5;
  int r0 = blockIdx.x * 64, n0 = blockIdx.y * 64;
  v8f zero = {0.f, 0.f, 0.f, 0.f, 0.f, 0.f, 0.f, 0.f};
  v8f acc[4];
#pragma unroll
  for (int i = 0; i < 4; ++i) acc[i] = zero;
  int rr = tid >> 1, ch = (tid & 1) * 16;
  const __bf16* arow = A + (size_t)(r0 + rr) * DD + ch;
  const __bf16* brow = Bw + (size_t)(n0 + rr) * DD + ch;

  auto stage = [&](int it, int par) {
    const __bf16* as = arow + it * 32;
    const __bf16* bs = brow + it * 32;
    unsigned la = (unsigned)(uintptr_t)(At[par] + rr * LDT + ch);
    unsigned lb = (unsigned)(uintptr_t)(Bt[par] + rr * LDT + ch);
    asm volatile("global_load_async_to_lds_b128 %0, %1, off" :: "v"(la), "v"(as) : "memory");
    asm volatile("global_load_async_to_lds_b128 %0, %1, off offset:16" :: "v"(la), "v"(as) : "memory");
    asm volatile("global_load_async_to_lds_b128 %0, %1, off" :: "v"(lb), "v"(bs) : "memory");
    asm volatile("global_load_async_to_lds_b128 %0, %1, off offset:16" :: "v"(lb), "v"(bs) : "memory");
  };

  stage(0, 0);
  asm volatile("s_wait_asynccnt 0" ::: "memory");
  __syncthreads();
  for (int it = 0; it < 24; ++it) {
    int par = it & 1;
    if (it + 1 < 24) stage(it + 1, par ^ 1);
    v16bf af = load_a_frag(At[par] + w * 16 * LDT, lane);
#pragma unroll
    for (int ns = 0; ns < 4; ++ns) {
      v16bf bf = load_b_frag(Bt[par] + ns * 16 * LDT, lane);
      acc[ns] = wmma_bf(af, bf, acc[ns]);
    }
    asm volatile("s_wait_asynccnt 0" ::: "memory");
    __syncthreads();
  }
  int hi = lane >> 4, col = lane & 15;
#pragma unroll
  for (int ns = 0; ns < 4; ++ns)
#pragma unroll
    for (int v = 0; v < 8; ++v) {
      int gr = r0 + w * 16 + hi * 8 + v;
      int gc = n0 + ns * 16 + col;
      Hout[(size_t)gr * HH2 + gc] = (__bf16)acc[ns][v];
    }
}

// ---------------- gated activation: act = y * silu(gate) ----------------
__global__ void act_kernel(const __bf16* __restrict__ h, __bf16* __restrict__ a) {
  size_t idx = (size_t)blockIdx.x * 256 + threadIdx.x;
  if (idx >= (size_t)RR * HH) return;
  size_t r = idx / HH, j = idx % HH;
  float y = (float)h[r * HH2 + j];
  float g = (float)h[r * HH2 + HH + j];
  float sg = 1.0f / (1.0f + __expf(-g));
  a[idx] = (__bf16)(y * g * sg);
}

// ---------------- fc2 GEMM via TDM tensor loads (TENSORcnt path) + residual ----------------
__global__ __launch_bounds__(128) void fc2_kernel(const __bf16* __restrict__ A,
                                                  const __bf16* __restrict__ Bw,  // (n,k) 768x3072
                                                  const float* __restrict__ res,
                                                  float* __restrict__ out) {
  __shared__ __bf16 At[2][TSZ];
  __shared__ __bf16 Bt[2][TSZ];
  int tid = threadIdx.x, lane = tid & 31, w = tid >> 5;
  int r0 = blockIdx.x * 64, n0 = blockIdx.y * 64;
  v8f zero = {0.f, 0.f, 0.f, 0.f, 0.f, 0.f, 0.f, 0.f};
  v8f acc[4];
#pragma unroll
  for (int i = 0; i < 4; ++i) acc[i] = zero;

  auto stage = [&](int it, int par) {
    tdm_load_64x32((unsigned)(uintptr_t)&At[par][0],
                   (const void*)(A + (size_t)r0 * HH + it * 32), HH);
    tdm_load_64x32((unsigned)(uintptr_t)&Bt[par][0],
                   (const void*)(Bw + (size_t)n0 * HH + it * 32), HH);
  };

  if (w == 0) {
    stage(0, 0);
    __builtin_amdgcn_s_wait_tensorcnt(0);
  }
  __syncthreads();
  for (int it = 0; it < 96; ++it) {
    int par = it & 1;
    if (w == 0 && it + 1 < 96) stage(it + 1, par ^ 1);
    v16bf af = load_a_frag(At[par] + w * 16 * LDT, lane);
#pragma unroll
    for (int ns = 0; ns < 4; ++ns) {
      v16bf bf = load_b_frag(Bt[par] + ns * 16 * LDT, lane);
      acc[ns] = wmma_bf(af, bf, acc[ns]);
    }
    if (w == 0 && it + 1 < 96) __builtin_amdgcn_s_wait_tensorcnt(0);
    __syncthreads();
  }
  int hi = lane >> 4, col = lane & 15;
#pragma unroll
  for (int ns = 0; ns < 4; ++ns)
#pragma unroll
    for (int v = 0; v < 8; ++v) {
      int gr = r0 + w * 16 + hi * 8 + v;
      int gc = n0 + ns * 16 + col;
      size_t o = (size_t)gr * DD + gc;
      out[o] = res[o] + acc[ns][v];
    }
}

// ---------------- host side ----------------
extern "C" void kernel_launch(void* const* d_in, const int* in_sizes, int n_in,
                              void* d_out, int out_size, void* d_ws, size_t ws_size,
                              hipStream_t stream) {
  (void)in_sizes; (void)n_in; (void)out_size; (void)ws_size;
  const float* x     = (const float*)d_in[0];
  const float* V     = (const float*)d_in[1];   // complex64 interleaved (NF, K)
  const float* sigma = (const float*)d_in[2];
  const float* Mu    = (const float*)d_in[3];
  const float* Mpp   = (const float*)d_in[4];
  const float* Mpm   = (const float*)d_in[5];
  const float* rn1w  = (const float*)d_in[6];
  const float* rn2w  = (const float*)d_in[7];
  const float* fc1w  = (const float*)d_in[8];
  const float* fc2w  = (const float*)d_in[9];

  char* ws = (char*)d_ws;
  float*  xn1f = (float*)(ws + OFS_XN1F);
  __bf16* xn1b = (__bf16*)(ws + OFS_XN1B);
  __bf16* U    = (__bf16*)(ws + OFS_U);
  float*  spec = (float*)(ws + OFS_SPEC);
  float*  out1 = (float*)(ws + OFS_OUT1);
  __bf16* xn2  = (__bf16*)(ws + OFS_XN2);
  __bf16* mpt  = (__bf16*)(ws + OFS_MPT);
  __bf16* mmt  = (__bf16*)(ws + OFS_MMT);
  __bf16* mut  = (__bf16*)(ws + OFS_MUT);
  __bf16* fc1b = (__bf16*)(ws + OFS_FC1B);   // (6144 x 768)
  __bf16* fc2b = (__bf16*)(ws + OFS_FC2B);   // (768 x 3072)
  float*  Z    = (float*)(ws + OFS_Z);
  __bf16* h    = (__bf16*)(ws + OFS_H);
  __bf16* actb = (__bf16*)(ws + OFS_ACT);
  float*  out  = (float*)d_out;

  {
    int nblk = (int)(((size_t)KK * DD * DD + 255) / 256);
    prep_phi_kernel<<<nblk, 256, 0, stream>>>(Mpp, sigma, mpt);
    prep_phi_kernel<<<nblk, 256, 0, stream>>>(Mpm, sigma, mmt);
  }
  prep_mu_kernel<<<(int)(((size_t)KUU * DD * DD + 255) / 256), 256, 0, stream>>>(Mu, mut);
  transpose_bf_kernel<<<dim3(HH2 / 32, DD / 32), 256, 0, stream>>>(fc1w, fc1b, DD, HH2);
  transpose_bf_kernel<<<dim3(DD / 32, HH / 32), 256, 0, stream>>>(fc2w, fc2b, HH, DD);

  // STU branch
  rmsnorm_kernel<<<RR, 256, 0, stream>>>(x, rn1w, xn1f, xn1b);
  fft_fwd_kernel<<<dim3(DD / 2, BB), 256, 0, stream>>>(xn1f, U);
  contract_kernel<<<dim3((NF + 63) / 64, DD / 64, BB * 2), 128, 0, stream>>>(U, mpt, mmt, V, Z);
  fft_inv_kernel<<<dim3(DD / 2, BB), 256, 0, stream>>>(Z, spec);
  ar_kernel<<<dim3(RR / 64, DD / 64), 128, 0, stream>>>(xn1b, mut, x, spec, out1);

  // MLP branch
  rmsnorm_kernel<<<RR, 256, 0, stream>>>(out1, rn2w, nullptr, xn2);
  fc1_kernel<<<dim3(RR / 64, HH2 / 64), 128, 0, stream>>>(xn2, fc1b, h);
  act_kernel<<<(int)(((size_t)RR * HH + 255) / 256), 256, 0, stream>>>(h, actb);
  fc2_kernel<<<dim3(RR / 64, DD / 64), 128, 0, stream>>>(actb, fc2b, out1, out);
}
